// HoME_86955907875074
// MI455X (gfx1250) — matrix operand order repaired and checked by reading, not compiled, not run on hardware
//
#include <hip/hip_runtime.h>

// ---------------------------------------------------------------------------
// HoME forward for MI455X (gfx1250): f16 WMMA GEMMs + small VALU kernels.
// All GEMMs: M%128==0, N%128==0, K%32==0 (true for every layer here).
// GEMM main loop: double-buffered LDS staged with GLOBAL_LOAD_ASYNC_TO_LDS
// (ASYNCcnt-tracked) when the toolchain exposes the builtins; otherwise a
// synchronous global->VGPR->LDS fallback.
// ---------------------------------------------------------------------------

typedef __attribute__((ext_vector_type(16))) _Float16 v16h;
typedef __attribute__((ext_vector_type(8)))  float    v8f;

union FragAB { v16h h; unsigned u[8]; };
union FragC  { v8f  f; float    x[8]; };

__device__ __forceinline__ float swishf(float x) { return x / (1.0f + __expf(-x)); }
__device__ __forceinline__ float sigf(float x)   { return 1.0f / (1.0f + __expf(-x)); }

#if defined(__has_builtin)
#if __has_builtin(__builtin_amdgcn_global_load_async_to_lds_b128) && \
    __has_builtin(__builtin_amdgcn_s_wait_asynccnt)
#define USE_ASYNC_LDS 1
#endif
#endif
#ifndef USE_ASYNC_LDS
#define USE_ASYNC_LDS 0
#endif

#if USE_ASYNC_LDS
// Builtin expects pointers to 128-bit int vectors (per compile diagnostic):
//   param0: __attribute__((vector_size(16))) int  __device__(AS1) *
typedef __attribute__((__vector_size__(4 * sizeof(int)))) int vb128_t;

__device__ __forceinline__ void async_copy16h(const _Float16* g, _Float16* l)
{
    // 8 halves = 128 bits per lane, memory -> LDS, tracked by ASYNCcnt.
    __builtin_amdgcn_global_load_async_to_lds_b128(
        (__attribute__((address_space(1))) vb128_t*)(void*)g,
        (__attribute__((address_space(3))) vb128_t*)(void*)l,
        0, 0);
}
#endif

// ---------------------------------------------------------------------------
// C[z] = act(A[z] * B[z] + bias[z])
// A: f16 [M,K] row-major, row stride lda, per-z offset strideA (elements)
// B: f16 [K,N] row-major (natural weight layout), per-z offset strideB
// C: f16, element (row,col) at C + z*strideC + row*ldc + col
// Block: 128x128 tile, 256 threads = 8 waves; wave grid 4(M) x 2(N),
// each wave: 2x4 tiles of 16x16, K stepped by 32 (one WMMA per tile per step).
// ---------------------------------------------------------------------------
__global__ __launch_bounds__(256)
void k_wmma_gemm(const _Float16* __restrict__ A, long long strideA, int lda,
                 const _Float16* __restrict__ Bm, long long strideB, int N,
                 const float* __restrict__ bias, long long strideBias,
                 _Float16* __restrict__ C, long long strideC, int ldc,
                 int M, int K, int act)
{
#if USE_ASYNC_LDS
    __shared__ _Float16 As[2][128 * 36];   // [buf][row][k] padded
    __shared__ _Float16 Bs[2][32 * 132];   // [buf][k][n]  padded
#else
    __shared__ _Float16 As[1][128 * 36];
    __shared__ _Float16 Bs[1][32 * 132];
#endif

    const int tid = threadIdx.x;
    const int z   = blockIdx.z;
    const _Float16* Az = A  + (long long)z * strideA;
    const _Float16* Bz = Bm + (long long)z * strideB;
    const float*    bz = bias + (long long)z * strideBias;

    const int bm = blockIdx.y * 128;
    const int bn = blockIdx.x * 128;

    // global->LDS assignments (16 halves per thread per tile)
    const int aRow = tid >> 1;          // 0..127
    const int aK   = (tid & 1) * 16;    // 0 or 16
    const int bRow = tid >> 3;          // 0..31 (k)
    const int bN   = (tid & 7) * 16;    // 0..112

    const int lane  = tid & 31;
    const int wave  = tid >> 5;
    const int wm    = (wave & 3) * 32;
    const int wn    = (wave >> 2) * 64;
    const int lrow  = lane & 15;
    const int khalf = (lane >> 4) * 8;  // 0 or 8

    FragC acc[2][4];
#pragma unroll
    for (int i = 0; i < 2; ++i)
#pragma unroll
        for (int j = 0; j < 4; ++j)
#pragma unroll
            for (int q = 0; q < 8; ++q) acc[i][j].x[q] = 0.0f;

    // fragment extraction + 8 WMMAs for one 32-deep K step from LDS buffers
    auto computeStep = [&](const _Float16* AsB, const _Float16* BsB) {
        FragAB fa[2], fb[4];
        // A fragment: lane -> row (M), VGPRs -> K pairs (khalf split per ISA)
#pragma unroll
        for (int mt = 0; mt < 2; ++mt) {
            const _Float16* p = &AsB[(wm + mt * 16 + lrow) * 36];
#pragma unroll
            for (int v = 0; v < 4; ++v) fa[mt].u[v]     = *(const unsigned*)(p + khalf + 2 * v);
#pragma unroll
            for (int v = 0; v < 4; ++v) fa[mt].u[4 + v] = *(const unsigned*)(p + 16 + khalf + 2 * v);
        }
        // B fragment: lane -> column (N), elements -> K (same khalf split as A)
#pragma unroll
        for (int nt = 0; nt < 4; ++nt) {
            const _Float16* p = &BsB[wn + nt * 16 + lrow];
#pragma unroll
            for (int s = 0; s < 16; ++s) fb[nt].h[s] = p[(s < 8 ? khalf + s : 16 + khalf + (s - 8)) * 132];
        }
#pragma unroll
        for (int mt = 0; mt < 2; ++mt)
#pragma unroll
            for (int nt = 0; nt < 4; ++nt)
                acc[mt][nt].f = __builtin_amdgcn_wmma_f32_16x16x32_f16(
                    false, fa[mt].h, false, fb[nt].h,
                    (short)0, acc[mt][nt].f, false, false);
    };

#if USE_ASYNC_LDS
    // Ping-pong double buffer fed by async memory->LDS DMA (4 b128 per wave
    // per tile pair); s_wait_asynccnt(4) keeps the next tile in flight while
    // guaranteeing the current tile has landed.
    auto issueTile = [&](int buf, int k0) {
        const _Float16* ga = Az + (long long)(bm + aRow) * lda + k0 + aK;
        _Float16* la = &As[buf][aRow * 36 + aK];
        async_copy16h(ga,     la);
        async_copy16h(ga + 8, la + 8);
        const _Float16* gb = Bz + (long long)(k0 + bRow) * N + bn + bN;
        _Float16* lb = &Bs[buf][bRow * 132 + bN];
        async_copy16h(gb,     lb);
        async_copy16h(gb + 8, lb + 8);
    };
    issueTile(0, 0);
    int cur = 0;
    for (int k0 = 0; k0 < K; k0 += 32) {
        if (k0 + 32 < K) {
            issueTile(cur ^ 1, k0 + 32);          // prefetch next tile
            __builtin_amdgcn_s_wait_asynccnt(4);  // current tile done, next in flight
        } else {
            __builtin_amdgcn_s_wait_asynccnt(0);
        }
        __syncthreads();                          // all waves' tiles visible
        computeStep(As[cur], Bs[cur]);
        __syncthreads();                          // done reading before refill
        cur ^= 1;
    }
#else
    for (int k0 = 0; k0 < K; k0 += 32) {
        __syncthreads();
        {   // A tile: 128 rows x 32 k
            const uint4* src = (const uint4*)(Az + (long long)(bm + aRow) * lda + k0 + aK);
            uint4 q0 = src[0], q1 = src[1];
            unsigned* dst = (unsigned*)&As[0][aRow * 36 + aK];
            dst[0] = q0.x; dst[1] = q0.y; dst[2] = q0.z; dst[3] = q0.w;
            dst[4] = q1.x; dst[5] = q1.y; dst[6] = q1.z; dst[7] = q1.w;
        }
        {   // B tile: 32 k x 128 n (K-major, matches lane<->K fragment layout)
            const uint4* src = (const uint4*)(Bz + (long long)(k0 + bRow) * N + bn + bN);
            uint4 q0 = src[0], q1 = src[1];
            unsigned* dst = (unsigned*)&Bs[0][bRow * 132 + bN];
            dst[0] = q0.x; dst[1] = q0.y; dst[2] = q0.z; dst[3] = q0.w;
            dst[4] = q1.x; dst[5] = q1.y; dst[6] = q1.z; dst[7] = q1.w;
        }
        __syncthreads();
        computeStep(As[0], Bs[0]);
    }
#endif

    // Epilogue: C/D layout lane -> N, VGPR i -> M row
    _Float16* Cz = C + (long long)z * strideC;
#pragma unroll
    for (int mt = 0; mt < 2; ++mt) {
        const int row0 = bm + wm + mt * 16 + (lane >> 4) * 8;
#pragma unroll
        for (int nt = 0; nt < 4; ++nt) {
            const int col = bn + wn + nt * 16 + lrow;
            const float bv = bz[col];
#pragma unroll
            for (int i = 0; i < 8; ++i) {
                float xv = acc[mt][nt].x[i] + bv;
                if (act) xv = swishf(xv);
                Cz[(long long)(row0 + i) * ldc + col] = (_Float16)xv;
            }
        }
    }
}

// ---------------------------------------------------------------------------
__global__ void k_convert(const float* __restrict__ s, _Float16* __restrict__ d, long long n)
{
    long long i = (long long)blockIdx.x * blockDim.x + threadIdx.x;
    if (i < n) d[i] = (_Float16)s[i];
}

// LayerNorm over last dim 256, in-place on f16, fp32 params. One wave per row.
__global__ __launch_bounds__(256)
void k_layernorm(_Float16* __restrict__ o, long long eStride, long long rowStride,
                 const float* __restrict__ g, const float* __restrict__ b,
                 long long pStride, int M)
{
    const int wave = threadIdx.x >> 5;
    const int lane = threadIdx.x & 31;
    const int row  = blockIdx.x * 8 + wave;
    if (row >= M) return;
    _Float16* p = o + (long long)blockIdx.y * eStride + (long long)row * rowStride;
    const float* gg = g + (long long)blockIdx.y * pStride;
    const float* bb = b + (long long)blockIdx.y * pStride;
    float x[8];
    float s = 0.f;
#pragma unroll
    for (int i = 0; i < 8; ++i) { x[i] = (float)p[lane + i * 32]; s += x[i]; }
#pragma unroll
    for (int m = 16; m >= 1; m >>= 1) s += __shfl_xor(s, m, 32);
    const float mu = s * (1.0f / 256.0f);
    float v = 0.f;
#pragma unroll
    for (int i = 0; i < 8; ++i) { float d0 = x[i] - mu; v += d0 * d0; }
#pragma unroll
    for (int m = 16; m >= 1; m >>= 1) v += __shfl_xor(v, m, 32);
    const float inv = rsqrtf(v * (1.0f / 256.0f) + 1e-5f);
#pragma unroll
    for (int i = 0; i < 8; ++i) {
        int k = lane + i * 32;
        p[k] = (_Float16)((x[i] - mu) * inv * gg[k] + bb[k]);
    }
}

// logits = h[m,:128] @ w2[128,NOUT] + b2 ; softmax over NOUT ; one thread per row
__global__ void k_gate_softmax(const _Float16* __restrict__ h, int ldh,
                               const float* __restrict__ w2, const float* __restrict__ b2,
                               float* __restrict__ out, int ldo, int M, int NOUT)
{
    int m = blockIdx.x * blockDim.x + threadIdx.x;
    if (m >= M) return;
    const _Float16* hp = h + (long long)m * ldh;
    float lg[8];
    float mx = -1e30f;
    for (int n = 0; n < NOUT; ++n) {
        float s = b2[n];
        for (int k = 0; k < 128; ++k) s += (float)hp[k] * w2[k * NOUT + n];
        lg[n] = s; mx = fmaxf(mx, s);
    }
    float sum = 0.f;
    for (int n = 0; n < NOUT; ++n) { lg[n] = __expf(lg[n] - mx); sum += lg[n]; }
    float inv = 1.0f / sum;
    for (int n = 0; n < NOUT; ++n) out[(long long)m * ldo + n] = lg[n] * inv;
}

// out[m,k] = sum_e o[e][m][k] * w[m, e]   (o: f16 [E][M][256])
__global__ void k_wsum(const _Float16* __restrict__ o, long long eStride, int E,
                       const float* __restrict__ w, int ldw,
                       float* __restrict__ out, long long MK)
{
    long long i = (long long)blockIdx.x * blockDim.x + threadIdx.x;
    if (i >= MK) return;
    long long m = i >> 8;
    float s = 0.f;
    for (int e = 0; e < E; ++e)
        s += (float)o[(long long)e * eStride + i] * w[m * ldw + e];
    out[i] = s;
}

// combined[b][t][0:256]=z_sh[b], combined[b][t][256:512]=z_gr[t>>1][b]
__global__ void k_build_combined(const float* __restrict__ z_sh, const float* __restrict__ z_gr,
                                 float* __restrict__ combined, int B)
{
    long long idx = (long long)blockIdx.x * blockDim.x + threadIdx.x;
    if (idx >= (long long)B * 2048) return;
    long long b = idx >> 11;
    int rem = (int)(idx & 2047);
    int t = rem >> 9;
    int d = rem & 511;
    float v;
    if (d < 256) v = z_sh[b * 256 + d];
    else         v = z_gr[(long long)(t >> 1) * B * 256 + b * 256 + (d - 256)];
    combined[idx] = v;
}

// lr[m,r] = sum_d combined[m,d] * fg_A[t,d,r]   (m = b*T+t, t = m&3)
__global__ void k_lora_r(const float* __restrict__ combined, const float* __restrict__ fgA,
                         float* __restrict__ lr, int MT)
{
    long long idx = (long long)blockIdx.x * blockDim.x + threadIdx.x;
    if (idx >= (long long)MT * 8) return;
    long long m = idx >> 3;
    int r = (int)(idx & 7);
    int t = (int)(m & 3);
    const float* x = combined + m * 512;
    const float* a = fgA + (long long)t * 4096 + r;
    float s = 0.f;
    for (int d = 0; d < 512; ++d) s += x[d] * a[d * 8];
    lr[idx] = s;
}

// combined[m,d] *= 2*sigmoid( sum_r lr[m,r]*fg_B[t,r,d] )
__global__ void k_lora_apply(float* __restrict__ combined, const float* __restrict__ fgB,
                             const float* __restrict__ lr, int MT)
{
    long long idx = (long long)blockIdx.x * blockDim.x + threadIdx.x;
    if (idx >= (long long)MT * 512) return;
    long long m = idx >> 9;
    int d = (int)(idx & 511);
    int t = (int)(m & 3);
    const float* bp = fgB + (long long)t * 4096 + d;
    const float* rp = lr + m * 8;
    float s = 0.f;
#pragma unroll
    for (int r = 0; r < 8; ++r) s += rp[r] * bp[r * 512];
    combined[idx] *= 2.0f * sigf(s);
}

// diag[m,e] = sum_k all_e[e][m][k] * sg_w[t,k,e] + sg_b[t,e]
__global__ void k_selfgate(const _Float16* __restrict__ all_e, const float* __restrict__ sg_w,
                           const float* __restrict__ sg_b, float* __restrict__ diag, int MT)
{
    long long idx = (long long)blockIdx.x * blockDim.x + threadIdx.x;
    if (idx >= (long long)MT * 8) return;
    int e = (int)(idx & 7);
    long long m = idx >> 3;
    int t = (int)(m & 3);
    const _Float16* x = all_e + (long long)e * MT * 256 + m * 256;
    const float* w = sg_w + (long long)t * 2048 + e;
    float s = sg_b[t * 8 + e];
    for (int k = 0; k < 256; ++k) s += (float)x[k] * w[k * 8];
    diag[idx] = s;
}

// agg[m,k] = sum_e all_e[e][m][k] * diag[m,e] * tg[m,e]
__global__ void k_agg(const _Float16* __restrict__ all_e, const float* __restrict__ diag,
                      const float* __restrict__ tg, float* __restrict__ agg, int MT)
{
    long long idx = (long long)blockIdx.x * blockDim.x + threadIdx.x;
    if (idx >= (long long)MT * 256) return;
    long long m = idx >> 8;
    float s = 0.f;
#pragma unroll
    for (int e = 0; e < 8; ++e)
        s += (float)all_e[(long long)e * MT * 256 + idx] * diag[m * 8 + e] * tg[m * 8 + e];
    agg[idx] = s;
}

// Tower 256->32(swish)->16(swish)->1(sigmoid). One wave per row, 8 rows/block.
__global__ __launch_bounds__(256)
void k_tower(const float* __restrict__ agg,
             const float* __restrict__ w1, const float* __restrict__ b1,
             const float* __restrict__ w2, const float* __restrict__ b2,
             const float* __restrict__ w3, const float* __restrict__ b3,
             float* __restrict__ out, int MT)
{
    __shared__ float hb[8][64];
    const int wave = threadIdx.x >> 5;
    const int lane = threadIdx.x & 31;
    const long long m = (long long)blockIdx.x * 8 + wave;
    const int t = (int)(m & 3);
    const float* x = agg + m * 256;

    float s = b1[t * 32 + lane];
    const float* W1 = w1 + (long long)t * 256 * 32 + lane;
    for (int k = 0; k < 256; ++k) s += x[k] * W1[k * 32];
    hb[wave][lane] = swishf(s);
    __syncthreads();
    if (lane < 16) {
        float s2 = b2[t * 16 + lane];
        const float* W2 = w2 + (long long)t * 32 * 16 + lane;
        for (int k = 0; k < 32; ++k) s2 += hb[wave][k] * W2[k * 16];
        hb[wave][32 + lane] = swishf(s2);
    }
    __syncthreads();
    if (lane == 0) {
        float s3 = b3[t];
        const float* W3 = w3 + t * 16;
        float acc = s3;
        for (int k = 0; k < 16; ++k) acc += hb[wave][32 + k] * W3[k];
        out[m] = sigf(acc);
    }
}

// ---------------------------------------------------------------------------
extern "C" void kernel_launch(void* const* d_in, const int* in_sizes, int n_in,
                              void* d_out, int out_size, void* d_ws, size_t ws_size,
                              hipStream_t stream)
{
    (void)in_sizes; (void)n_in; (void)out_size; (void)ws_size;

    // Inputs flattened in JAX pytree (sorted-key) order of setup_inputs():
    // params first (sorted), then v.
    enum {
        IN_FG_A = 0, IN_FG_B,
        IN_L1GR_B1, IN_L1GR_B2, IN_L1GR_B3, IN_L1GR_BETA, IN_L1GR_G, IN_L1GR_W1, IN_L1GR_W2, IN_L1GR_W3,
        IN_L1GRG_B1, IN_L1GRG_B2, IN_L1GRG_W1, IN_L1GRG_W2,
        IN_L1SH_B1, IN_L1SH_B2, IN_L1SH_B3, IN_L1SH_BETA, IN_L1SH_G, IN_L1SH_W1, IN_L1SH_W2, IN_L1SH_W3,
        IN_L1SHG_B1, IN_L1SHG_B2, IN_L1SHG_W1, IN_L1SHG_W2,
        IN_L2GR_B1, IN_L2GR_B2, IN_L2GR_B3, IN_L2GR_BETA, IN_L2GR_G, IN_L2GR_W1, IN_L2GR_W2, IN_L2GR_W3,
        IN_L2SH_B1, IN_L2SH_B2, IN_L2SH_B3, IN_L2SH_BETA, IN_L2SH_G, IN_L2SH_W1, IN_L2SH_W2, IN_L2SH_W3,
        IN_SG_B, IN_SG_W,
        IN_TG_B1, IN_TG_B2, IN_TG_W1, IN_TG_W2,
        IN_TW_B1, IN_TW_B2, IN_TW_B3, IN_TW_W1, IN_TW_W2, IN_TW_W3,
        IN_V
    };
    auto F = [&](int i) { return (const float*)d_in[i]; };

    const int B = 8192, MT = 32768;   // MT = B*T

    // ---- workspace bump allocator (256B aligned) ----
    char* base = (char*)d_ws;
    size_t off = 0;
    auto alloc = [&](size_t bytes) -> char* {
        char* p = base + off;
        off = (off + bytes + 255) & ~(size_t)255;
        return p;
    };

    // persistent
    _Float16* v16       = (_Float16*)alloc((size_t)B * 1024 * 2);
    _Float16* wf_l1sh1  = (_Float16*)alloc((size_t)4 * 1024 * 512 * 2);
    _Float16* wf_l1sh2  = (_Float16*)alloc((size_t)4 * 512 * 256 * 2);
    _Float16* wf_l1sh3  = (_Float16*)alloc((size_t)4 * 256 * 256 * 2);
    _Float16* wf_l1gr1  = (_Float16*)alloc((size_t)8 * 1024 * 512 * 2);
    _Float16* wf_l1gr2  = (_Float16*)alloc((size_t)8 * 512 * 256 * 2);
    _Float16* wf_l1gr3  = (_Float16*)alloc((size_t)8 * 256 * 256 * 2);
    _Float16* wf_shg1   = (_Float16*)alloc((size_t)1024 * 128 * 2);
    _Float16* wf_grg1   = (_Float16*)alloc((size_t)2 * 1024 * 128 * 2);
    _Float16* wf_l2sh1  = (_Float16*)alloc((size_t)4 * 512 * 512 * 2);
    _Float16* wf_l2sh2  = (_Float16*)alloc((size_t)4 * 512 * 256 * 2);
    _Float16* wf_l2sh3  = (_Float16*)alloc((size_t)4 * 256 * 256 * 2);
    _Float16* wf_l2gr1  = (_Float16*)alloc((size_t)8 * 512 * 512 * 2);
    _Float16* wf_l2gr2  = (_Float16*)alloc((size_t)8 * 512 * 256 * 2);
    _Float16* wf_l2gr3  = (_Float16*)alloc((size_t)8 * 256 * 256 * 2);
    _Float16* wf_tgg1   = (_Float16*)alloc((size_t)4 * 512 * 128 * 2);
    float*    combined  = (float*)   alloc((size_t)MT * 512 * 4);
    _Float16* combined16= (_Float16*)alloc((size_t)MT * 512 * 2);
    float*    tg        = (float*)   alloc((size_t)MT * 8 * 4);
    float*    diag      = (float*)   alloc((size_t)MT * 8 * 4);
    float*    aggb      = (float*)   alloc((size_t)MT * 256 * 4);
    float*    lora_r    = (float*)   alloc((size_t)MT * 8 * 4);
    float*    z_sh      = (float*)   alloc((size_t)B * 256 * 4);
    float*    z_gr      = (float*)   alloc((size_t)2 * B * 256 * 4);
    float*    gw        = (float*)   alloc((size_t)B * 4 * 4);
    float*    ggw       = (float*)   alloc((size_t)B * 8 * 4);
    _Float16* hg_sh     = (_Float16*)alloc((size_t)B * 128 * 2);
    _Float16* hg_gr     = (_Float16*)alloc((size_t)2 * B * 128 * 2);
    _Float16* h_tg      = (_Float16*)alloc((size_t)4 * B * 128 * 2);

    const size_t scratch0 = off;
    // layer-1 scratch
    _Float16* h1_sh = (_Float16*)alloc((size_t)4 * B * 512 * 2);
    _Float16* h2_sh = (_Float16*)alloc((size_t)4 * B * 256 * 2);
    _Float16* o_sh  = (_Float16*)alloc((size_t)4 * B * 256 * 2);
    _Float16* h1_gr = (_Float16*)alloc((size_t)8 * B * 512 * 2);
    _Float16* h2_gr = (_Float16*)alloc((size_t)8 * B * 256 * 2);
    _Float16* o_gr  = (_Float16*)alloc((size_t)8 * B * 256 * 2);

    // ---- helpers ----
    auto conv = [&](const float* s, _Float16* d, long long n) {
        k_convert<<<(unsigned)((n + 255) / 256), 256, 0, stream>>>(s, d, n);
    };
    auto gemm = [&](const _Float16* A, long long sA, int lda,
                    const _Float16* Bp, long long sB, int N,
                    const float* bias, long long sBias,
                    _Float16* C, long long sC, int ldc,
                    int M, int K, int Z, int act) {
        dim3 grid(N / 128, M / 128, Z);
        k_wmma_gemm<<<grid, 256, 0, stream>>>(A, sA, lda, Bp, sB, N, bias, sBias,
                                              C, sC, ldc, M, K, act);
    };

    // ---- fp32 -> f16 conversion pre-pass ----
    conv(F(IN_V),        v16,      (long long)B * 1024);
    conv(F(IN_L1SH_W1),  wf_l1sh1, 4LL * 1024 * 512);
    conv(F(IN_L1SH_W2),  wf_l1sh2, 4LL * 512 * 256);
    conv(F(IN_L1SH_W3),  wf_l1sh3, 4LL * 256 * 256);
    conv(F(IN_L1GR_W1),  wf_l1gr1, 8LL * 1024 * 512);
    conv(F(IN_L1GR_W2),  wf_l1gr2, 8LL * 512 * 256);
    conv(F(IN_L1GR_W3),  wf_l1gr3, 8LL * 256 * 256);
    conv(F(IN_L1SHG_W1), wf_shg1,  1024LL * 128);
    conv(F(IN_L1GRG_W1), wf_grg1,  2LL * 1024 * 128);
    conv(F(IN_L2SH_W1),  wf_l2sh1, 4LL * 512 * 512);
    conv(F(IN_L2SH_W2),  wf_l2sh2, 4LL * 512 * 256);
    conv(F(IN_L2SH_W3),  wf_l2sh3, 4LL * 256 * 256);
    conv(F(IN_L2GR_W1),  wf_l2gr1, 8LL * 512 * 512);
    conv(F(IN_L2GR_W2),  wf_l2gr2, 8LL * 512 * 256);
    conv(F(IN_L2GR_W3),  wf_l2gr3, 8LL * 256 * 256);
    conv(F(IN_TG_W1),    wf_tgg1,  4LL * 512 * 128);

    // ---- layer 1: shared expert bank (z = 4 experts) ----
    gemm(v16, 0, 1024, wf_l1sh1, 1024LL * 512, 512, F(IN_L1SH_B1), 512,
         h1_sh, (long long)B * 512, 512, B, 1024, 4, 1);
    gemm(h1_sh, (long long)B * 512, 512, wf_l1sh2, 512LL * 256, 256, F(IN_L1SH_B2), 256,
         h2_sh, (long long)B * 256, 256, B, 512, 4, 1);
    gemm(h2_sh, (long long)B * 256, 256, wf_l1sh3, 256LL * 256, 256, F(IN_L1SH_B3), 256,
         o_sh, (long long)B * 256, 256, B, 256, 4, 0);
    k_layernorm<<<dim3(B / 8, 4), 256, 0, stream>>>(o_sh, (long long)B * 256, 256,
                                                    F(IN_L1SH_G), F(IN_L1SH_BETA), 256, B);

    // ---- layer 1: group expert bank (z = 8 = 2 groups x 4 experts) ----
    gemm(v16, 0, 1024, wf_l1gr1, 1024LL * 512, 512, F(IN_L1GR_B1), 512,
         h1_gr, (long long)B * 512, 512, B, 1024, 8, 1);
    gemm(h1_gr, (long long)B * 512, 512, wf_l1gr2, 512LL * 256, 256, F(IN_L1GR_B2), 256,
         h2_gr, (long long)B * 256, 256, B, 512, 8, 1);
    gemm(h2_gr, (long long)B * 256, 256, wf_l1gr3, 256LL * 256, 256, F(IN_L1GR_B3), 256,
         o_gr, (long long)B * 256, 256, B, 256, 8, 0);
    k_layernorm<<<dim3(B / 8, 8), 256, 0, stream>>>(o_gr, (long long)B * 256, 256,
                                                    F(IN_L1GR_G), F(IN_L1GR_BETA), 256, B);

    // ---- layer 1 gates ----
    gemm(v16, 0, 1024, wf_shg1, 0, 128, F(IN_L1SHG_B1), 0,
         hg_sh, 0, 128, B, 1024, 1, 1);
    gemm(v16, 0, 1024, wf_grg1, 1024LL * 128, 128, F(IN_L1GRG_B1), 128,
         hg_gr, (long long)B * 128, 128, B, 1024, 2, 1);
    k_gate_softmax<<<B / 256, 256, 0, stream>>>(hg_sh, 128, F(IN_L1SHG_W2), F(IN_L1SHG_B2),
                                                gw, 4, B, 4);
    for (int g = 0; g < 2; ++g)
        k_gate_softmax<<<B / 256, 256, 0, stream>>>(hg_gr + (long long)g * B * 128, 128,
                                                    F(IN_L1GRG_W2) + g * 512, F(IN_L1GRG_B2) + g * 4,
                                                    ggw + g * 4, 8, B, 4);

    // ---- gate-weighted sums ----
    k_wsum<<<(unsigned)(((long long)B * 256 + 255) / 256), 256, 0, stream>>>(
        o_sh, (long long)B * 256, 4, gw, 4, z_sh, (long long)B * 256);
    for (int g = 0; g < 2; ++g)
        k_wsum<<<(unsigned)(((long long)B * 256 + 255) / 256), 256, 0, stream>>>(
            o_gr + (long long)g * 4 * B * 256, (long long)B * 256, 4, ggw + g * 4, 8,
            z_gr + (long long)g * B * 256, (long long)B * 256);

    // ---- combined + LoRA feature gate ----
    k_build_combined<<<(unsigned)(((long long)B * 2048 + 255) / 256), 256, 0, stream>>>(
        z_sh, z_gr, combined, B);
    k_lora_r<<<(unsigned)(((long long)MT * 8 + 255) / 256), 256, 0, stream>>>(
        combined, F(IN_FG_A), lora_r, MT);
    k_lora_apply<<<(unsigned)(((long long)MT * 512 + 255) / 256), 256, 0, stream>>>(
        combined, F(IN_FG_B), lora_r, MT);
    conv(combined, combined16, (long long)MT * 512);

    // ---- layer 2 scratch (reuse layer-1 scratch region) ----
    off = scratch0;
    _Float16* all_e = (_Float16*)alloc((size_t)8 * MT * 256 * 2);  // [slot 0..7][m][256]
    _Float16* h1_2  = (_Float16*)alloc((size_t)4 * MT * 512 * 2);
    _Float16* h2_2  = (_Float16*)alloc((size_t)4 * MT * 256 * 2);

    // ---- layer 2: shared bank (M = B*T, z = 4) -> all_e slots 0..3 ----
    gemm(combined16, 0, 512, wf_l2sh1, 512LL * 512, 512, F(IN_L2SH_B1), 512,
         h1_2, (long long)MT * 512, 512, MT, 512, 4, 1);
    gemm(h1_2, (long long)MT * 512, 512, wf_l2sh2, 512LL * 256, 256, F(IN_L2SH_B2), 256,
         h2_2, (long long)MT * 256, 256, MT, 512, 4, 1);
    gemm(h2_2, (long long)MT * 256, 256, wf_l2sh3, 256LL * 256, 256, F(IN_L2SH_B3), 256,
         all_e, (long long)MT * 256, 256, MT, 256, 4, 0);
    k_layernorm<<<dim3(MT / 8, 4), 256, 0, stream>>>(all_e, (long long)MT * 256, 256,
                                                     F(IN_L2SH_G), F(IN_L2SH_BETA), 256, MT);

    // ---- layer 2: group bank per task (rows strided by T) -> all_e slots 4..7 ----
    for (int t = 0; t < 4; ++t) {
        const int g = t >> 1;   // GROUP_MAP = [0,0,1,1]
        gemm(combined16 + t * 512, 0, 2048,
             wf_l2gr1 + (long long)g * 4 * 512 * 512, 512LL * 512, 512,
             F(IN_L2GR_B1) + (long long)g * 4 * 512, 512,
             h1_2, (long long)B * 512, 512, B, 512, 4, 1);
        gemm(h1_2, (long long)B * 512, 512,
             wf_l2gr2 + (long long)g * 4 * 512 * 256, 512LL * 256, 256,
             F(IN_L2GR_B2) + g * 1024, 256,
             h2_2, (long long)B * 256, 256, B, 512, 4, 1);
        gemm(h2_2, (long long)B * 256, 256,
             wf_l2gr3 + (long long)g * 4 * 256 * 256, 256LL * 256, 256,
             F(IN_L2GR_B3) + g * 1024, 256,
             all_e + 4LL * MT * 256 + t * 256, (long long)MT * 256, 1024, B, 256, 4, 0);
        k_layernorm<<<dim3(B / 8, 4), 256, 0, stream>>>(
            all_e + 4LL * MT * 256 + t * 256, (long long)MT * 256, 1024,
            F(IN_L2GR_G) + g * 1024, F(IN_L2GR_BETA) + g * 1024, 256, B);
    }

    // ---- task gate (z = task) ----
    gemm(combined16, 512, 2048, wf_tgg1, 512LL * 128, 128, F(IN_TG_B1), 128,
         h_tg, (long long)B * 128, 128, B, 512, 4, 1);
    for (int t = 0; t < 4; ++t)
        k_gate_softmax<<<B / 256, 256, 0, stream>>>(h_tg + (long long)t * B * 128, 128,
                                                    F(IN_TG_W2) + t * 1024, F(IN_TG_B2) + t * 8,
                                                    tg + t * 8, 32, B, 8);

    // ---- self-gate diag, aggregate, towers ----
    k_selfgate<<<(unsigned)(((long long)MT * 8 + 255) / 256), 256, 0, stream>>>(
        all_e, F(IN_SG_W), F(IN_SG_B), diag, MT);
    k_agg<<<(unsigned)(((long long)MT * 256 + 255) / 256), 256, 0, stream>>>(
        all_e, diag, tg, aggb, MT);
    k_tower<<<MT / 8, 256, 0, stream>>>(aggb,
                                        F(IN_TW_W1), F(IN_TW_B1),
                                        F(IN_TW_W2), F(IN_TW_B2),
                                        F(IN_TW_W3), F(IN_TW_B3),
                                        (float*)d_out, MT);
}
